// RecurrentNeuralNetworkDecoder_57526791963205
// MI455X (gfx1250) — matrix operand order repaired and checked by reading, not compiled, not run on hardware
//
#include <hip/hip_runtime.h>
#include <hip/hip_bf16.h>

typedef __attribute__((ext_vector_type(16))) __bf16         v16bf;
typedef __attribute__((ext_vector_type(8)))  float          v8f;
typedef __attribute__((ext_vector_type(8)))  unsigned short v8u;
typedef __attribute__((ext_vector_type(4)))  float          v4f;

#define B_ 64
#define S_ 512
#define D_ 1024
#define H_ 1024
#define O_ 1024

__device__ __forceinline__ unsigned short f2bf(float f) {
  unsigned int u = __builtin_bit_cast(unsigned int, f);
  u += 0x7FFFu + ((u >> 16) & 1u);          // round-to-nearest-even
  return (unsigned short)(u >> 16);
}

union AB {
  unsigned short u[16];
  v8u h[2];
  v16bf v;
};

// Pack W [K x N] row-major f32 -> bf16 B-fragments, tile-major:
// dst[((nt*KT + kt)*32 + lane)*16 + i] =
//     bf16(W[(kt*32 + (lane&16 ? 8 : 0) + i + (i >= 8 ? 8 : 0))*N + nt*16 + (lane&15)])
// So in the GEMM each lane loads its whole 16-elem fragment as 32 contiguous bytes.
__global__ void pack_w(const float* __restrict__ W, unsigned short* __restrict__ dst,
                       int K, int N) {
  int tid  = blockIdx.x * blockDim.x + threadIdx.x;
  int KT   = K >> 5;
  int lane = tid & 31;
  int tile = tid >> 5;
  int total = (N >> 4) * KT;
  if (tile >= total) return;
  int nt = tile / KT, kt = tile - nt * KT;
  int n  = nt * 16 + (lane & 15);
  int kb = kt * 32 + ((lane & 16) ? 8 : 0);
  AB f;
#pragma unroll
  for (int i = 0; i < 16; ++i) f.u[i] = f2bf(W[(size_t)(kb + i + (i >= 8 ? 8 : 0)) * N + n]);
  v8u* out = (v8u*)(dst + ((size_t)tile * 32 + lane) * 16);
  out[0] = f.h[0];
  out[1] = f.h[1];
}

// One GRU time step. Grid: 64 blocks (one per 16-column tile of H).
// Block: 768 threads = 24 waves = (3 gates) x (4 row tiles) x (2 K-halves).
// K-half 0 covers K 0..1023 (the x part), K-half 1 covers K 1024..2047 (the h part).
__global__ __launch_bounds__(768) void gru_step(
    const float* __restrict__ x, const float* __restrict__ bg,
    const unsigned short* __restrict__ Wg_pack,
    float* __restrict__ h_f32,
    const unsigned short* __restrict__ h_in,   // bf16 h_{t-1}  [64,1024]
    unsigned short* __restrict__ h_out,        // bf16 h_t      [64,1024]
    unsigned short* __restrict__ hs,           // bf16 [B,S,H]
    int t) {
  __shared__ float gbuf[2][3][64][16];         // [khalf][gate][b][col] partial gates

  const int nt   = blockIdx.x;                 // H column tile 0..63
  const int tidx = threadIdx.x;
  const int wave = tidx >> 5;
  const int lane = tidx & 31;
  const int half = wave & 1;                   // K half
  const int pair = wave >> 1;
  const int g    = pair >> 2;                  // gate: 0=r 1=z 2=n
  const int m    = pair & 3;                   // row tile (batch/16)
  const int row  = m * 16 + (lane & 15);       // batch index b
  const int koff = (lane & 16) ? 8 : 0;        // A-fragment lane-group K offset

  const int nt_total = g * 64 + nt;            // column tile in 3H space
  const unsigned short* Bbase =
      Wg_pack + (size_t)(nt_total * 64 + half * 32) * (32 * 16);

  v8f acc = {};
  if (half == 0) {
    const float* xrow = x + ((size_t)row * S_ + t) * D_;
    for (int kt = 0; kt < 32; ++kt) {
      v16bf bfrag = *(const v16bf*)(Bbase + ((size_t)kt * 32 + lane) * 16);
      AB a;
      int k0 = kt * 32 + koff;
      const v4f* p0 = (const v4f*)(xrow + k0);
      const v4f* p1 = (const v4f*)(xrow + k0 + 16);
      v4f xa = p0[0], xb = p0[1], xc = p1[0], xd = p1[1];
#pragma unroll
      for (int i = 0; i < 4; ++i) {
        a.u[i]      = f2bf(xa[i]);
        a.u[4 + i]  = f2bf(xb[i]);
        a.u[8 + i]  = f2bf(xc[i]);
        a.u[12 + i] = f2bf(xd[i]);
      }
      acc = __builtin_amdgcn_wmma_f32_16x16x32_bf16(false, a.v, false, bfrag,
                                                    (short)0, acc, false, false);
    }
    float bias = bg[g * H_ + nt * 16 + (lane & 15)];   // bias added by half 0 only
#pragma unroll
    for (int r = 0; r < 8; ++r) acc[r] += bias;
  } else {
    const unsigned short* hrow = h_in + (size_t)row * H_;
    for (int kt = 0; kt < 32; ++kt) {
      v16bf bfrag = *(const v16bf*)(Bbase + ((size_t)kt * 32 + lane) * 16);
      AB a;
      int k0 = kt * 32 + koff;
      a.h[0] = *(const v8u*)(hrow + k0);
      a.h[1] = *(const v8u*)(hrow + k0 + 16);
      acc = __builtin_amdgcn_wmma_f32_16x16x32_bf16(false, a.v, false, bfrag,
                                                    (short)0, acc, false, false);
    }
  }

  {   // spill C tile to LDS: lane l holds N=l%16, rows r + 8*(l>=16)
    int mbase = m * 16 + ((lane & 16) ? 8 : 0);
    int col   = lane & 15;
#pragma unroll
    for (int r = 0; r < 8; ++r) gbuf[half][g][mbase + r][col] = acc[r];
  }
  __syncthreads();

  // GRU elementwise update for this block's 16 hidden columns (exclusive owner).
  for (int idx = tidx; idx < 64 * 16; idx += 768) {
    int b  = idx >> 4, jj = idx & 15;
    int col = nt * 16 + jj;
    float rg = gbuf[0][0][b][jj] + gbuf[1][0][b][jj];
    float zg = gbuf[0][1][b][jj] + gbuf[1][1][b][jj];
    float ng = gbuf[0][2][b][jj] + gbuf[1][2][b][jj];
    float hp = h_f32[(size_t)b * H_ + col];
    float r  = 1.f / (1.f + __expf(-rg));
    float z  = 1.f / (1.f + __expf(-zg));
    float n  = tanhf(ng + r * hp);
    float hn = (1.f - z) * n + z * hp;
    h_f32[(size_t)b * H_ + col] = hn;
    unsigned short hb = f2bf(hn);
    h_out[(size_t)b * H_ + col] = hb;
    hs[((size_t)b * S_ + t) * H_ + col] = hb;
  }
}

// out[32768,1024] = hs[32768,1024] @ Wc[1024,1024] + bc.  8 waves/block, 1 tile/wave.
__global__ __launch_bounds__(256) void proj(
    const unsigned short* __restrict__ hs, const unsigned short* __restrict__ Wc_pack,
    const float* __restrict__ bc, float* __restrict__ out) {
  const int lane = threadIdx.x & 31;
  const int wave = threadIdx.x >> 5;
  const int tile = blockIdx.x * 8 + wave;      // 2048*64 = 131072 tiles
  const int mt   = tile >> 6;
  const int ntc  = tile & 63;
  const int row  = mt * 16 + (lane & 15);
  const int koff = (lane & 16) ? 8 : 0;
  const unsigned short* arow  = hs + (size_t)row * H_;
  const unsigned short* Bbase = Wc_pack + (size_t)ntc * 32 * (32 * 16);

  v8f acc = {};
  for (int kt = 0; kt < 32; ++kt) {
    v16bf bfrag = *(const v16bf*)(Bbase + ((size_t)kt * 32 + lane) * 16);
    AB a;
    int k0 = kt * 32 + koff;
    a.h[0] = *(const v8u*)(arow + k0);
    a.h[1] = *(const v8u*)(arow + k0 + 16);
    acc = __builtin_amdgcn_wmma_f32_16x16x32_bf16(false, a.v, false, bfrag,
                                                  (short)0, acc, false, false);
  }
  float bias = bc[ntc * 16 + (lane & 15)];
  int mbase = mt * 16 + ((lane & 16) ? 8 : 0);
  int col   = ntc * 16 + (lane & 15);
#pragma unroll
  for (int r = 0; r < 8; ++r)
    out[(size_t)(mbase + r) * O_ + col] = acc[r] + bias;
}

extern "C" void kernel_launch(void* const* d_in, const int* in_sizes, int n_in,
                              void* d_out, int out_size, void* d_ws, size_t ws_size,
                              hipStream_t stream) {
  const float* x  = (const float*)d_in[0];   // [B,S,D]
  const float* Wg = (const float*)d_in[1];   // [2048,3072]
  const float* bg = (const float*)d_in[2];   // [3072]
  const float* Wc = (const float*)d_in[3];   // [1024,1024]
  const float* bc = (const float*)d_in[4];   // [1024]
  float* out = (float*)d_out;                // [B,S,O]

  char* ws = (char*)d_ws;
  size_t o = 0;
  unsigned short* Wg_pack = (unsigned short*)(ws + o); o += (size_t)2048 * 3072 * 2;
  unsigned short* Wc_pack = (unsigned short*)(ws + o); o += (size_t)1024 * 1024 * 2;
  float*          h_f32   = (float*)(ws + o);          o += (size_t)64 * 1024 * 4;
  unsigned short* h_bf    = (unsigned short*)(ws + o); o += (size_t)2 * 64 * 1024 * 2;
  unsigned short* hs      = (unsigned short*)(ws + o); o += (size_t)64 * 512 * 1024 * 2;

  // one-time per call: pack weights to bf16 fragments; zero hidden state
  pack_w<<<(192 * 64 * 32 + 255) / 256, 256, 0, stream>>>(Wg, Wg_pack, 2048, 3072);
  pack_w<<<( 64 * 32 * 32 + 255) / 256, 256, 0, stream>>>(Wc, Wc_pack, 1024, 1024);
  hipMemsetAsync(h_f32, 0, (size_t)64 * 1024 * 4, stream);
  hipMemsetAsync(h_bf,  0, (size_t)2 * 64 * 1024 * 2, stream);

  // sequential recurrence: one launch per step (kernel boundary = global sync)
  for (int t = 0; t < 512; ++t) {
    const unsigned short* h_in  = h_bf + (size_t)(t & 1) * 64 * 1024;
    unsigned short*       h_out = h_bf + (size_t)((t + 1) & 1) * 64 * 1024;
    gru_step<<<64, 768, 0, stream>>>(x, bg, Wg_pack, h_f32, h_in, h_out, hs, t);
  }

  proj<<<16384, 256, 0, stream>>>(hs, Wc_pack, bc, out);
}
